// MultiheadAttention_25074019074635
// MI455X (gfx1250) — matrix operand order repaired and checked by reading, not compiled, not run on hardware
//
#include <hip/hip_runtime.h>
#include <hip/hip_bf16.h>
#include <stdint.h>

// ---------------------------------------------------------------------------
// MHA forward for S=2048, B=2, E=1024, H=16, D=64 on gfx1250 (wave32, WMMA).
// Pipeline: cvt(bf16) -> QKV GEMM (WMMA + async-to-LDS double buffer)
//           -> RoPE/pack -> flash attention (WMMA) -> out-proj GEMM.
// ---------------------------------------------------------------------------

typedef __attribute__((ext_vector_type(16))) __bf16 v16bf;
typedef __attribute__((ext_vector_type(8)))  __bf16 v8bf;
typedef __attribute__((ext_vector_type(8)))  float  v8f;

#define SEQ  2048
#define BB   2
#define EE   1024
#define HH   16
#define DD   64
#define MROWS (SEQ * BB)   // 4096

__device__ __forceinline__ __bf16 f2bf(float f) {
  union { float fv; unsigned u; } a; a.fv = f;
  unsigned r = a.u + 0x7FFFu + ((a.u >> 16) & 1u);   // round-to-nearest-even
  union { unsigned short s; __bf16 b; } o; o.s = (unsigned short)(r >> 16);
  return o.b;
}

__device__ __forceinline__ v8f wmma_bf16(v16bf a, v16bf b, v8f c) {
  // D(f32 16x16) = A(bf16 16x32) * B(bf16 32x16) + C
  return __builtin_amdgcn_wmma_f32_16x16x32_bf16(
      /*neg_a=*/false, a, /*neg_b=*/false, b,
      /*c_mod=*/(short)0, c, /*reuse_a=*/false, /*reuse_b=*/false);
}

#define SHUF16(lo, hi) __builtin_shufflevector((lo), (hi), \
    0,1,2,3,4,5,6,7,8,9,10,11,12,13,14,15)

// gfx1250 async global->LDS copy (ASYNCcnt), GVS addressing:
//   lds[vdst + ...] = mem[saddr + vaddr]
__device__ __forceinline__ void async_ld_lds_b128(unsigned lds_byte_addr,
                                                  unsigned gvoff_bytes,
                                                  uint64_t sbase) {
  asm volatile("global_load_async_to_lds_b128 %0, %1, %2"
               :: "v"(lds_byte_addr), "v"(gvoff_bytes), "s"(sbase)
               : "memory");
}
__device__ __forceinline__ void wait_async0() {
  asm volatile("s_wait_asynccnt 0x0" ::: "memory");
}

// ---------------------------------------------------------------------------
// fp32 -> bf16 elementwise convert
// ---------------------------------------------------------------------------
__global__ void cvt_f32_bf16(const float* __restrict__ in,
                             __bf16* __restrict__ out, int n) {
  int i = blockIdx.x * blockDim.x + threadIdx.x;
  if (i < n) out[i] = f2bf(in[i]);
}

// ---------------------------------------------------------------------------
// Tiled WMMA GEMM: out[M,N] = A[M,K](bf16) * W[N,K]^T(bf16) + bias
// 256 threads = 8 waves; block tile M=256 (32 rows/wave), N=64; K step 32.
// W tile double-buffered in LDS via global_load_async_to_lds_b128.
// ---------------------------------------------------------------------------
__global__ __launch_bounds__(256)
void gemm_bf16_wmma(const __bf16* __restrict__ A, const __bf16* __restrict__ W,
                    const float* __restrict__ bias, float* __restrict__ out,
                    int M, int N, int K) {
  __shared__ __attribute__((aligned(32))) __bf16 wtile[2][64 * 32]; // 2 x 4KB
  const int tid  = threadIdx.x;
  const int lane = tid & 31;
  const int wave = tid >> 5;
  const int hi   = lane >> 4;      // 0: lanes 0-15, 1: lanes 16-31
  const int l16  = lane & 15;
  const int m0   = blockIdx.x * 256 + wave * 32;
  const int n0   = blockIdx.y * 64;

  // per-thread 16B chunk of the staged W tile
  const int trow = tid >> 2;               // 0..63   (n_local)
  const int tcc  = (tid & 3) * 8;          // 0,8,16,24 (k elems)
  const unsigned ldsBuf0 = (unsigned)(size_t)(&wtile[0][0]);
  const unsigned ldsOff  = (unsigned)(trow * 32 + tcc) * 2u;
  const unsigned bufStep = (unsigned)(64 * 32 * 2);      // 4096 bytes
  const uint64_t wBase   = (uint64_t)(uintptr_t)W;
  const size_t   gElem   = (size_t)(n0 + trow) * K + tcc; // + kk per step

  v8f acc[2][4] = {};              // 32x64 output per wave

  const int nk = K / 32;
  // prefetch K-step 0 into buffer 0
  async_ld_lds_b128(ldsBuf0 + ldsOff, (unsigned)(gElem * 2), wBase);

  for (int ik = 0; ik < nk; ++ik) {
    const int p  = ik & 1;
    const int kk = ik * 32;
    wait_async0();                 // my slice of buffer p landed in LDS
    __syncthreads();               // everyone's slice landed; prev reads done
    if (ik + 1 < nk) {             // prefetch next tile into the other buffer
      async_ld_lds_b128(ldsBuf0 + (unsigned)(1 - p) * bufStep + ldsOff,
                        (unsigned)((gElem + (size_t)(ik + 1) * 32) * 2), wBase);
    }

    // A fragments for two 16-row subtiles (16x32 each)
    v16bf afrag[2];
    #pragma unroll
    for (int ms = 0; ms < 2; ++ms) {
      const __bf16* ar = &A[(size_t)(m0 + ms * 16 + l16) * K + kk];
      v8bf lo = *(const v8bf*)&ar[hi * 16];
      v8bf h8 = *(const v8bf*)&ar[hi * 16 + 8];
      afrag[ms] = SHUF16(lo, h8);
    }

    #pragma unroll
    for (int ns = 0; ns < 4; ++ns) {
      v16bf bfrag = *(const v16bf*)&wtile[p][(ns * 16 + l16) * 32 + hi * 16];
      acc[0][ns] = wmma_bf16(afrag[0], bfrag, acc[0][ns]);
      acc[1][ns] = wmma_bf16(afrag[1], bfrag, acc[1][ns]);
    }
  }

  // epilogue: C/D layout row m = r + 8*hi, col n = l16 (+16 per subtile)
  #pragma unroll
  for (int ms = 0; ms < 2; ++ms) {
    #pragma unroll
    for (int ns = 0; ns < 4; ++ns) {
      int n = n0 + ns * 16 + l16;
      float b = bias ? bias[n] : 0.0f;
      #pragma unroll
      for (int r = 0; r < 8; ++r) {
        int m = m0 + ms * 16 + r + 8 * hi;
        out[(size_t)m * N + n] = acc[ms][ns][r] + b;
      }
    }
  }
}

// ---------------------------------------------------------------------------
// RoPE (interleaved) + layout pack.
//   proj  : [S,B,3E] f32 (QKV fused)
//   Qm,Km : [B,H,S,D] bf16  (Q pre-scaled by 1/sqrt(D))
//   Vt    : [B,H,D,S] bf16  (transposed so V B-fragments load contiguously)
// ---------------------------------------------------------------------------
__global__ void rope_pack(const float* __restrict__ proj,
                          const float* __restrict__ sin_t,
                          const float* __restrict__ cos_t,
                          __bf16* __restrict__ Qm, __bf16* __restrict__ Km,
                          __bf16* __restrict__ Vt) {
  int idx = blockIdx.x * blockDim.x + threadIdx.x;   // S*B*H*(D/2) = 2M
  const int DP = DD / 2;
  int p = idx % DP; int t = idx / DP;
  int h = t % HH;   t /= HH;
  int b = t % BB;   int s = t / BB;
  if (s >= SEQ) return;

  size_t row = ((size_t)s * BB + b) * (3 * EE);
  int e = h * DD + p * 2;
  float q0 = proj[row + e],            q1 = proj[row + e + 1];
  float k0 = proj[row + EE + e],       k1 = proj[row + EE + e + 1];
  float v0 = proj[row + 2 * EE + e],   v1 = proj[row + 2 * EE + e + 1];
  float sn = sin_t[s * DP + p], cs = cos_t[s * DP + p];

  // interleaved RoPE: out[2i] = x0*cos - x1*sin ; out[2i+1] = x1*cos + x0*sin
  float qo0 = q0 * cs - q1 * sn, qo1 = q1 * cs + q0 * sn;
  float ko0 = k0 * cs - k1 * sn, ko1 = k1 * cs + k0 * sn;
  const float qscale = 0.125f;   // 1/sqrt(64)

  size_t bh = (size_t)b * HH + h;
  size_t qk = (bh * SEQ + s) * DD + p * 2;
  Qm[qk]     = f2bf(qo0 * qscale);
  Qm[qk + 1] = f2bf(qo1 * qscale);
  Km[qk]     = f2bf(ko0);
  Km[qk + 1] = f2bf(ko1);
  size_t vb = bh * DD * SEQ;
  Vt[vb + (size_t)(p * 2) * SEQ + s]     = f2bf(v0);
  Vt[vb + (size_t)(p * 2 + 1) * SEQ + s] = f2bf(v1);
}

// ---------------------------------------------------------------------------
// Causal flash attention. Grid: (B*H) * (S/64) workgroups; 128 threads
// (4 waves), one wave per 16 query rows. Key blocks of 32.
//   Qm,Km: [B,H,S,D] bf16 ; Vt: [B,H,D,S] bf16 ; Oc: [S,B,E] bf16
// ---------------------------------------------------------------------------
__global__ __launch_bounds__(128)
void flash_attn(const __bf16* __restrict__ Qm, const __bf16* __restrict__ Km,
                const __bf16* __restrict__ Vt, __bf16* __restrict__ Oc) {
  __shared__ __attribute__((aligned(32))) __bf16 pstage[4][16 * 32]; // P per wave

  int bh   = blockIdx.x >> 5;          // B*H = 32
  int qblk = blockIdx.x & 31;          // S/64 = 32
  int lane = threadIdx.x & 31;
  int wave = threadIdx.x >> 5;
  int hi = lane >> 4, l16 = lane & 15;
  int qb = qblk * 64 + wave * 16;

  const __bf16* Qbh = Qm + (size_t)bh * SEQ * DD;
  const __bf16* Kbh = Km + (size_t)bh * SEQ * DD;
  const __bf16* Vbh = Vt + (size_t)bh * DD * SEQ;

  // Q A-fragments for D chunks [0,32) and [32,64)
  v16bf qfrag[2];
  {
    const __bf16* qrow = &Qbh[(size_t)(qb + l16) * DD];
    #pragma unroll
    for (int db = 0; db < 2; ++db) {
      v8bf lo = *(const v8bf*)&qrow[db * 32 + hi * 16];
      v8bf h8 = *(const v8bf*)&qrow[db * 32 + hi * 16 + 8];
      qfrag[db] = SHUF16(lo, h8);
    }
  }

  v8f o[4] = {};                 // O accumulators: 16 rows x 64 dims
  float mrow[8], lrow[8];        // per-lane stats for rows r + 8*hi
  #pragma unroll
  for (int r = 0; r < 8; ++r) { mrow[r] = -1e30f; lrow[r] = 0.0f; }

  const int kend = qb + 16;      // keys 0..qb+15 needed (causal)
  for (int kb = 0; kb < kend; kb += 32) {
    // ---- S = Q @ K^T over two 16-key tiles -------------------------------
    v8f s0 = {}, s1 = {};
    #pragma unroll
    for (int db = 0; db < 2; ++db) {
      v16bf b0 = *(const v16bf*)&Kbh[(size_t)(kb + l16) * DD + db * 32 + hi * 16];
      s0 = wmma_bf16(qfrag[db], b0, s0);
      v16bf b1 = *(const v16bf*)&Kbh[(size_t)(kb + 16 + l16) * DD + db * 32 + hi * 16];
      s1 = wmma_bf16(qfrag[db], b1, s1);
    }

    // ---- causal mask + online softmax ------------------------------------
    int qrow0 = qb + 8 * hi;
    #pragma unroll
    for (int r = 0; r < 8; ++r) {
      int qi = qrow0 + r;
      s0[r] = (kb + l16      > qi) ? -1e30f : s0[r];
      s1[r] = (kb + 16 + l16 > qi) ? -1e30f : s1[r];
    }
    #pragma unroll
    for (int r = 0; r < 8; ++r) {
      float mx = fmaxf(s0[r], s1[r]);
      #pragma unroll
      for (int off = 1; off < 16; off <<= 1)
        mx = fmaxf(mx, __shfl_xor(mx, off, 32));
      float mnew = fmaxf(mrow[r], mx);
      float corr = __expf(mrow[r] - mnew);
      float p0 = __expf(s0[r] - mnew);
      float p1 = __expf(s1[r] - mnew);
      s0[r] = p0; s1[r] = p1;
      float psum = p0 + p1;
      #pragma unroll
      for (int off = 1; off < 16; off <<= 1)
        psum += __shfl_xor(psum, off, 32);
      lrow[r] = lrow[r] * corr + psum;
      mrow[r] = mnew;
      o[0][r] *= corr; o[1][r] *= corr; o[2][r] *= corr; o[3][r] *= corr;
    }

    // ---- transpose P (C/D layout -> A layout) through LDS ----------------
    __bf16* ps = pstage[wave];
    #pragma unroll
    for (int r = 0; r < 8; ++r) {
      int m = r + 8 * hi;
      ps[m * 32 + l16]      = f2bf(s0[r]);
      ps[m * 32 + 16 + l16] = f2bf(s1[r]);
    }
    v16bf pfrag;
    {
      const __bf16* prow = &ps[l16 * 32];
      v8bf lo = *(const v8bf*)&prow[hi * 16];
      v8bf h8 = *(const v8bf*)&prow[hi * 16 + 8];
      pfrag = SHUF16(lo, h8);
    }

    // ---- O += P @ V (V from transposed layout: contiguous keys) ----------
    #pragma unroll
    for (int nt = 0; nt < 4; ++nt) {
      v16bf vfrag = *(const v16bf*)&Vbh[(size_t)(nt * 16 + l16) * SEQ + kb + hi * 16];
      o[nt] = wmma_bf16(pfrag, vfrag, o[nt]);
    }
  }

  // ---- normalize and write to [S,B,E] (E = h*64 + d) ---------------------
  int b = bh >> 4, h = bh & 15;
  #pragma unroll
  for (int r = 0; r < 8; ++r) {
    int m = r + 8 * hi;
    int s = qb + m;
    float inv = 1.0f / lrow[r];
    #pragma unroll
    for (int nt = 0; nt < 4; ++nt) {
      int d = nt * 16 + l16;
      Oc[((size_t)s * BB + b) * EE + h * DD + d] = f2bf(o[nt][r] * inv);
    }
  }
}

// ---------------------------------------------------------------------------
// Host-side launch
// ---------------------------------------------------------------------------
extern "C" void kernel_launch(void* const* d_in, const int* in_sizes, int n_in,
                              void* d_out, int out_size, void* d_ws, size_t ws_size,
                              hipStream_t stream) {
  const float* query = (const float*)d_in[0];
  // d_in[1] = attn_mask (unused: causal)
  const float* sin_t = (const float*)d_in[2];
  const float* cos_t = (const float*)d_in[3];
  const float* w_in  = (const float*)d_in[4];
  const float* b_in  = (const float*)d_in[5];
  const float* w_out = (const float*)d_in[6];
  const float* b_out = (const float*)d_in[7];
  float* out = (float*)d_out;

  char* ws = (char*)d_ws;
  __bf16* Xbf  = (__bf16*)(ws);                          //  8 MB  query bf16 [4096,1024]
  __bf16* Wqkv = (__bf16*)(ws + ( 8ull << 20));          //  6 MB  [3072,1024]
  __bf16* Wout = (__bf16*)(ws + (14ull << 20));          //  2 MB  [1024,1024]
  float*  proj = (float* )(ws + (16ull << 20));          // 48 MB  [4096,3072]
  __bf16* Qm   = (__bf16*)(ws + (64ull << 20));          //  8 MB  [B,H,S,D]
  __bf16* Km   = (__bf16*)(ws + (72ull << 20));          //  8 MB  [B,H,S,D]
  __bf16* Vt   = (__bf16*)(ws + (80ull << 20));          //  8 MB  [B,H,D,S]
  __bf16* Oc   = (__bf16*)(ws + (88ull << 20));          //  8 MB  [4096,1024]

  // 1) fp32 -> bf16 converts
  cvt_f32_bf16<<<(MROWS * EE + 255) / 256, 256, 0, stream>>>(query, Xbf, MROWS * EE);
  cvt_f32_bf16<<<(3 * EE * EE + 255) / 256, 256, 0, stream>>>(w_in, Wqkv, 3 * EE * EE);
  cvt_f32_bf16<<<(EE * EE + 255) / 256, 256, 0, stream>>>(w_out, Wout, EE * EE);

  // 2) QKV projection: proj = X @ Wqkv^T + b_in   (4096 x 3072 x 1024)
  dim3 g1(MROWS / 256, (3 * EE) / 64);
  gemm_bf16_wmma<<<g1, 256, 0, stream>>>(Xbf, Wqkv, b_in, proj, MROWS, 3 * EE, EE);

  // 3) RoPE + pack Q/K/V
  rope_pack<<<(SEQ * BB * HH * (DD / 2) + 255) / 256, 256, 0, stream>>>(
      proj, sin_t, cos_t, Qm, Km, Vt);

  // 4) causal flash attention
  flash_attn<<<BB * HH * (SEQ / 64), 128, 0, stream>>>(Qm, Km, Vt, Oc);

  // 5) output projection: out = Oc @ Wout^T + b_out   (4096 x 1024 x 1024)
  dim3 g2(MROWS / 256, EE / 64);
  gemm_bf16_wmma<<<g2, 256, 0, stream>>>(Oc, Wout, b_out, out, MROWS, EE, EE);
}
